// baseline_snn_64132451664225
// MI455X (gfx1250) — matrix operand order repaired and checked by reading, not compiled, not run on hardware
//
#include <hip/hip_runtime.h>

typedef __attribute__((ext_vector_type(16))) _Float16 v16h;
typedef __attribute__((ext_vector_type(8)))  _Float16 v8h;
typedef __attribute__((ext_vector_type(8)))  float    v8f;

#define DECAY_M_F 0.7788007830714049f   // exp(-1/4)
#define THRESH_F  1.0f

#define BATCH 256
#define T_LEN 100
#define CIN   784
#define HID   500
#define NOUT  10
#define K1P   800   // 784 padded up to multiple of 32
#define NP    512   // 500 padded up to multiple of 64
#define N3P   16    // 10 padded to one WMMA tile
#define MBT   (BATCH * T_LEN)  // 25600 flattened (b,t) rows

// ---------------------------------------------------------------------------
// Pack W[o, k] (f32, row-major O x Cin) into per-fragment-contiguous f16 B:
// fragment layout (ISA 7.12.2, 16-bit B 32x16, wave32):
//   lane l<16 : halves 0..7 = B[k0+0..7,  n0+l],    8..15 = B[k0+16..23, n0+l]
//   lane l>=16: halves 0..7 = B[k0+8..15, n0+l-16], 8..15 = B[k0+24..31, n0+l-16]
// Packed element index: idx = ((kt*ntiles + nt)*32 + lane)*16 + j
// ---------------------------------------------------------------------------
__global__ void pack_w(const float* __restrict__ W, _Float16* __restrict__ Bp,
                       int O, int Cin, int Kpad, int Npad) {
  int idx = blockIdx.x * blockDim.x + threadIdx.x;
  int total = Kpad * Npad;
  if (idx >= total) return;
  int j    = idx & 15;
  int lane = (idx >> 4) & 31;
  int tile = idx >> 9;
  int ntiles = Npad >> 4;
  int kt = tile / ntiles;
  int nt = tile - kt * ntiles;
  int k = kt * 32 + ((j >> 3) << 4) + ((lane >> 4) << 3) + (j & 7);
  int n = nt * 16 + (lane & 15);
  float val = (k < Cin && n < O) ? W[(size_t)n * Cin + k] : 0.0f;
  Bp[idx] = (_Float16)val;
}

// f32 [rows x cols] -> f16 [rows x Kpad] row-major, zero pad
__global__ void pack_a(const float* __restrict__ X, _Float16* __restrict__ A,
                       int rows, int cols, int Kpad) {
  int idx = blockIdx.x * blockDim.x + threadIdx.x;
  if (idx >= rows * Kpad) return;
  int r = idx / Kpad;
  int c = idx - r * Kpad;
  A[idx] = (_Float16)((c < cols) ? X[(size_t)r * cols + c] : 0.0f);
}

// ---------------------------------------------------------------------------
// WMMA GEMM: D[M x Npad] f32 = A[M x Kpad] f16 (row-major) * Bpacked.
// 128 threads = 4 waves. Each wave owns 32 rows; block covers 128 rows x
// (16*NT) cols. The B k-slice (NT KB) is shared by all 4 waves: it is staged
// into LDS with async global->LDS copies (ASYNCcnt) and double-buffered; the
// compute reads it back with ds_load_b128 while the next slice streams in.
// All NT B fragments are pulled into VGPRs first (one DScnt wait), then the
// 2*NT WMMAs issue back-to-back (disjoint accumulators -> no hazard NOPs).
// A fragment (ISA 7.12.2, 16-bit A 16x32):
//   lane l<16 : halves 0..7 = A[m0+l, k0+0..7],  8..15 = A[m0+l, k0+16..23]
//   lane l>=16: halves 0..7 = A[m0+l-16, k0+8..15], 8..15 = k0+24..31
// ---------------------------------------------------------------------------
template <int NT>
__global__ __launch_bounds__(128) void gemm_wmma(
    const _Float16* __restrict__ A, const _Float16* __restrict__ Bp,
    float* __restrict__ D, int M, int Npad, int Kpad) {
  const int lane = threadIdx.x & 31;
  const int wave = threadIdx.x >> 5;
  const int m0 = (blockIdx.y * 4 + wave) * 32;
  const int n0 = blockIdx.x * (16 * NT);
  const int ntiles = Npad >> 4;
  const int nt0 = n0 >> 4;
  const int KT = Kpad >> 5;

  __shared__ __align__(32) _Float16 bsh[2][NT * 32 * 16];

  v8f acc0[NT] = {};
  v8f acc1[NT] = {};

  const int row  = m0 + (lane & 15);
  const int koff = (lane >> 4) << 3;  // 0 or 8
  const _Float16* aptr0 = A + (size_t)row * Kpad + koff;
  const _Float16* aptr1 = aptr0 + (size_t)16 * Kpad;

  const int  c       = threadIdx.x;       // B chunk id: tile = c>>5, lane = c&31
  const bool bloader = (c < NT * 32);

  // stage first B k-slice (async global -> LDS, no VGPR round-trip)
  if (bloader) {
    const _Float16* gp =
        Bp + (((size_t)(nt0 + (c >> 5)) * 32 + (c & 31)) << 4);
    unsigned la = (unsigned)(size_t)&bsh[0][(size_t)c << 4];
    asm volatile("global_load_async_to_lds_b128 %0, %1, off"
                 :: "v"(la), "v"((unsigned long long)(size_t)gp) : "memory");
  }
  asm volatile("s_wait_asynccnt 0x0" ::: "memory");
  __syncthreads();

  for (int kt = 0; kt < KT; ++kt) {
    const int cur = kt & 1;
    // prefetch next B k-slice into the other LDS buffer
    if (kt + 1 < KT && bloader) {
      const _Float16* gp =
          Bp + (((size_t)((kt + 1) * ntiles + nt0 + (c >> 5)) * 32 + (c & 31)) << 4);
      unsigned la = (unsigned)(size_t)&bsh[1 - cur][(size_t)c << 4];
      asm volatile("global_load_async_to_lds_b128 %0, %1, off"
                   :: "v"(la), "v"((unsigned long long)(size_t)gp) : "memory");
    }

    const int k0 = kt << 5;
    __builtin_prefetch(aptr0 + k0 + 32, 0, 3);
    v8h a0lo = *(const v8h*)(aptr0 + k0);
    v8h a0hi = *(const v8h*)(aptr0 + k0 + 16);
    v8h a1lo = *(const v8h*)(aptr1 + k0);
    v8h a1hi = *(const v8h*)(aptr1 + k0 + 16);
    v16h af0, af1;
#pragma unroll
    for (int i = 0; i < 8; ++i) {
      af0[i] = a0lo[i]; af0[i + 8] = a0hi[i];
      af1[i] = a1lo[i]; af1[i + 8] = a1hi[i];
    }

    // gather all B fragments first: one DScnt wait instead of NT
    v16h bf[NT];
#pragma unroll
    for (int j = 0; j < NT; ++j)
      bf[j] = *(const v16h*)&bsh[cur][(size_t)((j * 32 + lane) << 4)];

#pragma unroll
    for (int j = 0; j < NT; ++j) {
      acc0[j] = __builtin_amdgcn_wmma_f32_16x16x32_f16(
          false, af0, false, bf[j], (short)0, acc0[j], false, false);
      acc1[j] = __builtin_amdgcn_wmma_f32_16x16x32_f16(
          false, af1, false, bf[j], (short)0, acc1[j], false, false);
    }

    // make prefetched slice visible; barrier also protects buf reuse
    asm volatile("s_wait_asynccnt 0x0" ::: "memory");
    __syncthreads();
  }

  // C/D layout: VGPR r, lanes 0-15 -> M=r, N=lane; lanes 16-31 -> M=r+8
  const int crow  = m0 + ((lane >> 4) << 3);
  const int ccol0 = n0 + (lane & 15);
#pragma unroll
  for (int j = 0; j < NT; ++j)
#pragma unroll
    for (int r = 0; r < 8; ++r) {
      D[(size_t)(crow + r) * Npad + ccol0 + j * 16]      = acc0[j][r];
      D[(size_t)(crow + 16 + r) * Npad + ccol0 + j * 16] = acc1[j][r];
    }
}

// ---------------------------------------------------------------------------
// Layer-1 LIF fused with layer-2 input IIR.
// I1[b,o,t] = P[b,o]*g[t] + b1[o], g[t] = 1 + c1*g[t-1] + c2*g[t-2]
// Writes A2 f16 at [(b*T+t)*NP + o]; pad columns o>=HID get zeros.
// ---------------------------------------------------------------------------
__global__ void lif1_iir2(const float* __restrict__ P, const float* __restrict__ b1,
                          const float* __restrict__ c1_1, const float* __restrict__ c2_1,
                          const float* __restrict__ c1_2, const float* __restrict__ c2_2,
                          _Float16* __restrict__ A2) {
  const int tid = blockIdx.x * blockDim.x + threadIdx.x;
  const int o = tid & (NP - 1);
  const int b = tid >> 9;
  if (b >= BATCH) return;
  _Float16* out = A2 + (size_t)b * T_LEN * NP + o;
  if (o >= HID) {
    for (int t = 0; t < T_LEN; ++t) out[(size_t)t * NP] = (_Float16)0.0f;
    return;
  }
  const float gc1 = c1_1[0], gc2 = c2_1[0];
  const float yc1 = c1_2[o], yc2 = c2_2[o];
  const float p = P[b * NP + o];
  const float bias = b1[o];
  float g1 = 0.f, g2 = 0.f, v = 0.f, s = 0.f, y1 = 0.f, y2 = 0.f;
  for (int t = 0; t < T_LEN; ++t) {
    float g = 1.0f + gc1 * g1 + gc2 * g2;
    g2 = g1; g1 = g;
    v = DECAY_M_F * v * (1.0f - s) + (p * g + bias);
    s = (v > THRESH_F) ? 1.0f : 0.0f;
    float y = s + yc1 * y1 + yc2 * y2;
    y2 = y1; y1 = y;
    out[(size_t)t * NP] = (_Float16)y;
  }
}

// Layer-2 LIF fused with layer-3 input IIR.
__global__ void lif2_iir3(const float* __restrict__ I2, const float* __restrict__ b2,
                          const float* __restrict__ c1_3, const float* __restrict__ c2_3,
                          _Float16* __restrict__ A3) {
  const int tid = blockIdx.x * blockDim.x + threadIdx.x;
  const int o = tid & (NP - 1);
  const int b = tid >> 9;
  if (b >= BATCH) return;
  _Float16* out = A3 + (size_t)b * T_LEN * NP + o;
  if (o >= HID) {
    for (int t = 0; t < T_LEN; ++t) out[(size_t)t * NP] = (_Float16)0.0f;
    return;
  }
  const float yc1 = c1_3[o], yc2 = c2_3[o];
  const float bias = b2[o];
  const float* iin = I2 + (size_t)b * T_LEN * NP + o;
  float v = 0.f, s = 0.f, y1 = 0.f, y2 = 0.f;
  for (int t = 0; t < T_LEN; ++t) {
    v = DECAY_M_F * v * (1.0f - s) + (iin[(size_t)t * NP] + bias);
    s = (v > THRESH_F) ? 1.0f : 0.0f;
    float y = s + yc1 * y1 + yc2 * y2;
    y2 = y1; y1 = y;
    out[(size_t)t * NP] = (_Float16)y;
  }
}

// Layer-3 LIF -> final spikes, reference layout [B, 10, T] row-major f32.
__global__ void lif3(const float* __restrict__ I3, const float* __restrict__ b3,
                     float* __restrict__ out) {
  const int tid = blockIdx.x * blockDim.x + threadIdx.x;
  const int o = tid & (N3P - 1);
  const int b = tid >> 4;
  if (b >= BATCH || o >= NOUT) return;
  const float bias = b3[o];
  const float* iin = I3 + (size_t)b * T_LEN * N3P + o;
  float* op = out + ((size_t)b * NOUT + o) * T_LEN;
  float v = 0.f, s = 0.f;
  for (int t = 0; t < T_LEN; ++t) {
    v = DECAY_M_F * v * (1.0f - s) + (iin[(size_t)t * N3P] + bias);
    s = (v > THRESH_F) ? 1.0f : 0.0f;
    op[t] = s;
  }
}

extern "C" void kernel_launch(void* const* d_in, const int* in_sizes, int n_in,
                              void* d_out, int out_size, void* d_ws, size_t ws_size,
                              hipStream_t stream) {
  const float* inputs = (const float*)d_in[0];
  const float* c1_1 = (const float*)d_in[1];
  const float* c2_1 = (const float*)d_in[2];
  const float* W1   = (const float*)d_in[3];
  const float* b1   = (const float*)d_in[4];
  const float* c1_2 = (const float*)d_in[5];
  const float* c2_2 = (const float*)d_in[6];
  const float* W2   = (const float*)d_in[7];
  const float* b2   = (const float*)d_in[8];
  const float* c1_3 = (const float*)d_in[9];
  const float* c2_3 = (const float*)d_in[10];
  const float* W3   = (const float*)d_in[11];
  const float* b3   = (const float*)d_in[12];

  char* ws = (char*)d_ws;
  size_t off = 0;
  auto alloc = [&](size_t bytes) {
    void* p = ws + off;
    off = (off + bytes + 255) & ~(size_t)255;
    return p;
  };
  _Float16* Bw1 = (_Float16*)alloc((size_t)K1P * NP * 2);
  _Float16* Bw2 = (_Float16*)alloc((size_t)NP * NP * 2);
  _Float16* Bw3 = (_Float16*)alloc((size_t)NP * N3P * 2);
  _Float16* A1  = (_Float16*)alloc((size_t)BATCH * K1P * 2);
  float*    Pb  = (float*)   alloc((size_t)BATCH * NP * 4);
  _Float16* A2  = (_Float16*)alloc((size_t)MBT * NP * 2);
  float*    I2  = (float*)   alloc((size_t)MBT * NP * 4);
  float*    I3  = (float*)   alloc((size_t)MBT * N3P * 4);
  _Float16* A3  = A2;  // GEMM2 consumed A2 before lif2 writes A3

  const int bs = 256;
  pack_w<<<(K1P * NP + bs - 1) / bs, bs, 0, stream>>>(W1, Bw1, HID, CIN, K1P, NP);
  pack_w<<<(NP * NP + bs - 1) / bs, bs, 0, stream>>>(W2, Bw2, HID, HID, NP, NP);
  pack_w<<<(NP * N3P + bs - 1) / bs, bs, 0, stream>>>(W3, Bw3, NOUT, HID, NP, N3P);
  pack_a<<<(BATCH * K1P + bs - 1) / bs, bs, 0, stream>>>(inputs, A1, BATCH, CIN, K1P);

  // Layer 1: single GEMM (time-separable input): 256 x 800 x 512
  gemm_wmma<4><<<dim3(NP / 64, BATCH / 128), 128, 0, stream>>>(A1, Bw1, Pb, BATCH, NP, K1P);
  lif1_iir2<<<(BATCH * NP) / bs, bs, 0, stream>>>(Pb, b1, c1_1, c2_1, c1_2, c2_2, A2);

  // Layer 2: dominant GEMM, 25600 x 512 x 512
  gemm_wmma<4><<<dim3(NP / 64, MBT / 128), 128, 0, stream>>>(A2, Bw2, I2, MBT, NP, NP);
  lif2_iir3<<<(BATCH * NP) / bs, bs, 0, stream>>>(I2, b2, c1_3, c2_3, A3);

  // Layer 3: 25600 x 512 x 16
  gemm_wmma<1><<<dim3(1, MBT / 128), 128, 0, stream>>>(A3, Bw3, I3, MBT, N3P, NP);
  lif3<<<(BATCH * N3P) / bs, bs, 0, stream>>>(I3, b3, (float*)d_out);
}
